// LSTM_Multivariate_8950711845786
// MI455X (gfx1250) — matrix operand order repaired and checked by reading, not compiled, not run on hardware
//
#include <hip/hip_runtime.h>
#include <hip/hip_bf16.h>
#include <math.h>

// ---------------------------------------------------------------------------
// LSTM (38 -> 64 -> 64 -> 38), T = 131072 steps.
//   Kernel 1 (parallel, WMMA): pre1[t, 0:256] = x[t] @ Wih1^T + (bih1 + bhh1)
//       -> big [T x 38] x [38 x 256] GEMM via v_wmma_f32_16x16x32_f16.
//   Kernel 2 (persistent single workgroup): sequential scan over t with all
//       recurrent weights staged in LDS (~279 KB of the 320 KB/WGP).
// ---------------------------------------------------------------------------

typedef __attribute__((ext_vector_type(16))) _Float16 v16h;
typedef __attribute__((ext_vector_type(8)))  _Float16 v8h;
typedef __attribute__((ext_vector_type(8)))  float    v8f;

#define FEATS 38
#define HID   64
#define G1    (4 * HID)    // 256
#define G3    (4 * FEATS)  // 152

// ======================== Kernel 1: input projection ========================
// One block = 16 timesteps; 8 waves x 2 column-tiles = 256 gate columns.
__global__ __launch_bounds__(256) void pre1_gemm_wmma(
    const float* __restrict__ x,      // [T, 38]
    const float* __restrict__ Wih1,   // [256, 38]
    const float* __restrict__ bih1,   // [256]
    const float* __restrict__ bhh1,   // [256]
    float* __restrict__ pre1,         // [T, 256]
    int T)
{
    __shared__ _Float16 xlds[16][64];    // K padded 38 -> 64 with zeros
    __shared__ _Float16 wlds[G1][64];
    __shared__ float    bsum[G1];

    const int tid = threadIdx.x;
    const int t0  = blockIdx.x * 16;

    for (int idx = tid; idx < G1 * 64; idx += 256) {
        const int row = idx >> 6, k = idx & 63;
        wlds[row][k] = (k < FEATS) ? (_Float16)Wih1[row * FEATS + k] : (_Float16)0.0f;
    }
    for (int idx = tid; idx < 16 * 64; idx += 256) {
        const int row = idx >> 6, k = idx & 63;
        const int t = t0 + row;
        xlds[row][k] = (k < FEATS && t < T) ? (_Float16)x[(size_t)t * FEATS + k]
                                            : (_Float16)0.0f;
    }
    bsum[tid] = bih1[tid] + bhh1[tid];
    __syncthreads();

    const int lane = tid & 31;
    const int wave = tid >> 5;
    const int row  = lane & 15;   // A: M row, B: N column, C: N column
    const int hi   = lane >> 4;   // half-wave select per ISA VGPR layout

    // A fragments (16x32 f16): lane holds K = hi*8..+7 and 16+hi*8..+7
    v16h a0, a1;
    {
        v8h l0 = *(const v8h*)&xlds[row][hi * 8];
        v8h u0 = *(const v8h*)&xlds[row][16 + hi * 8];
        v8h l1 = *(const v8h*)&xlds[row][32 + hi * 8];
        v8h u1 = *(const v8h*)&xlds[row][48 + hi * 8];
#pragma unroll
        for (int i = 0; i < 8; ++i) {
            a0[i] = l0[i]; a0[8 + i] = u0[i];
            a1[i] = l1[i]; a1[8 + i] = u1[i];
        }
    }

    const bool full_tile = (t0 + 16 <= T);   // block-uniform: no exec churn

#pragma unroll
    for (int s = 0; s < 2; ++s) {
        const int jbase = (wave * 2 + s) * 16;
        const int j     = jbase + row;
        // B fragments (32x16 f16): lane holds K = hi*8..+7 and 16+hi*8..+7
        v16h b0 = __builtin_shufflevector(*(const v8h*)&wlds[j][hi * 8], *(const v8h*)&wlds[j][16 + hi * 8], 0,1,2,3,4,5,6,7,8,9,10,11,12,13,14,15);
        v16h b1 = __builtin_shufflevector(*(const v8h*)&wlds[j][32 + hi * 8], *(const v8h*)&wlds[j][48 + hi * 8], 0,1,2,3,4,5,6,7,8,9,10,11,12,13,14,15);
        v8f c = {};
        c = __builtin_amdgcn_wmma_f32_16x16x32_f16(false, a0, false, b0,
                                                   (short)0, c, false, false);
        c = __builtin_amdgcn_wmma_f32_16x16x32_f16(false, a1, false, b1,
                                                   (short)0, c, false, false);
        const float bv = bsum[j];
        // C layout: VGPR r -> M = r + 8*hi, N = lane&15
        if (full_tile) {
#pragma unroll
            for (int r = 0; r < 8; ++r) {
                pre1[(size_t)(t0 + r + hi * 8) * G1 + j] = c[r] + bv;
            }
        } else {
#pragma unroll
            for (int r = 0; r < 8; ++r) {
                const int t = t0 + r + hi * 8;
                if (t < T) pre1[(size_t)t * G1 + j] = c[r] + bv;
            }
        }
    }
}

// ======================== Kernel 2: sequential scan =========================
// LDS layout (float offsets). Weight rows padded to stride 68 (=4 mod 64 in
// dwords -> at most 2-way bank conflict across 32 lanes, b128-aligned).
constexpr int OFF_WHH1 = 0;                        // 256 x 68
constexpr int OFF_WIH2 = OFF_WHH1 + G1 * 68;       // 256 x 68
constexpr int OFF_WHH2 = OFF_WIH2 + G1 * 68;       // 256 x 68
constexpr int OFF_WIH3 = OFF_WHH2 + G1 * 68;       // 152 x 68
constexpr int OFF_WHH3 = OFF_WIH3 + G3 * 68;       // 152 x 40 (K 38 -> 40, zero pad)
constexpr int OFF_B2   = OFF_WHH3 + G3 * 40;       // 256
constexpr int OFF_B3   = OFF_B2 + G1;              // 160 (152 used)
constexpr int OFF_H1   = OFF_B3 + 160;             // 64
constexpr int OFF_C1   = OFF_H1 + HID;
constexpr int OFF_H2   = OFF_C1 + HID;
constexpr int OFF_C2   = OFF_H2 + HID;
constexpr int OFF_H3   = OFF_C2 + HID;             // 40 (38 + zero pad)
constexpr int OFF_C3   = OFF_H3 + 40;              // 40
constexpr int OFF_Z    = OFF_C3 + 40;              // 256
constexpr int LDS_FLOATS = OFF_Z + G1;             // 69648 floats = 278592 B

__device__ __forceinline__ float sigm(float x) { return 1.0f / (1.0f + __expf(-x)); }
__device__ __forceinline__ float tanh_fast(float x) {
    const float e = __expf(2.0f * x);     // inf-safe: e=inf -> 1, e=0 -> -1
    return 1.0f - 2.0f / (e + 1.0f);
}

template <int NVEC>
__device__ __forceinline__ float dot_row(const float* __restrict__ w,
                                         const float* __restrict__ h) {
    const float4* wv = (const float4*)w;
    const float4* hv = (const float4*)h;
    float a0 = 0.f, a1 = 0.f, a2 = 0.f, a3 = 0.f;
#pragma unroll
    for (int k = 0; k < NVEC; ++k) {
        const float4 a = wv[k];
        const float4 b = hv[k];   // broadcast read (all lanes same address)
        a0 = fmaf(a.x, b.x, a0); a1 = fmaf(a.y, b.y, a1);
        a2 = fmaf(a.z, b.z, a2); a3 = fmaf(a.w, b.w, a3);
    }
    return (a0 + a1) + (a2 + a3);
}

__global__ __launch_bounds__(256) void lstm_seq(
    const float* __restrict__ pre1,   // [T, 256]
    const float* __restrict__ h1_in, const float* __restrict__ c1_in,
    const float* __restrict__ h2_in, const float* __restrict__ c2_in,
    const float* __restrict__ h3_in, const float* __restrict__ c3_in,
    const float* __restrict__ Whh1,   // [256, 64]
    const float* __restrict__ Wih2,   // [256, 64]
    const float* __restrict__ Whh2,   // [256, 64]
    const float* __restrict__ bih2, const float* __restrict__ bhh2,
    const float* __restrict__ Wih3,   // [152, 64]
    const float* __restrict__ Whh3,   // [152, 38]
    const float* __restrict__ bih3, const float* __restrict__ bhh3,
    float* __restrict__ out,          // [T, 38]
    int T)
{
    extern __shared__ float lds[];
    const int tid = threadIdx.x;

    // ---- stage weights / biases / state into LDS (once) ----
    for (int idx = tid; idx < G1 * HID; idx += 256) {
        const int r = idx >> 6, k = idx & 63;
        lds[OFF_WHH1 + r * 68 + k] = Whh1[idx];
        lds[OFF_WIH2 + r * 68 + k] = Wih2[idx];
        lds[OFF_WHH2 + r * 68 + k] = Whh2[idx];
    }
    for (int idx = tid; idx < G3 * HID; idx += 256) {
        lds[OFF_WIH3 + (idx >> 6) * 68 + (idx & 63)] = Wih3[idx];
    }
    for (int idx = tid; idx < G3 * 40; idx += 256) {
        const int r = idx / 40, k = idx - r * 40;
        lds[OFF_WHH3 + idx] = (k < FEATS) ? Whh3[r * FEATS + k] : 0.0f;
    }
    lds[OFF_B2 + tid] = bih2[tid] + bhh2[tid];
    if (tid < G3) lds[OFF_B3 + tid] = bih3[tid] + bhh3[tid];
    if (tid < HID) {
        lds[OFF_H1 + tid] = h1_in[tid]; lds[OFF_C1 + tid] = c1_in[tid];
        lds[OFF_H2 + tid] = h2_in[tid]; lds[OFF_C2 + tid] = c2_in[tid];
    }
    if (tid < 40) {
        lds[OFF_H3 + tid] = (tid < FEATS) ? h3_in[tid] : 0.0f;
        lds[OFF_C3 + tid] = (tid < FEATS) ? c3_in[tid] : 0.0f;
    }
    __syncthreads();

    // ---- sequential scan ----
    for (int t = 0; t < T; ++t) {
        if (t + 16 < T)  // stream next pre-activations toward the WGP
            __builtin_prefetch(&pre1[(size_t)(t + 16) * G1 + tid], 0, 0);

        // layer 1: z[j] = pre1[t][j] + Whh1[j,:] . h1
        {
            float acc = pre1[(size_t)t * G1 + tid];
            acc += dot_row<16>(&lds[OFF_WHH1 + tid * 68], &lds[OFF_H1]);
            lds[OFF_Z + tid] = acc;
        }
        __syncthreads();
        if (tid < HID) {
            const float i = sigm(lds[OFF_Z + tid]);
            const float f = sigm(lds[OFF_Z + HID + tid]);
            const float g = tanh_fast(lds[OFF_Z + 2 * HID + tid]);
            const float o = sigm(lds[OFF_Z + 3 * HID + tid]);
            const float c = fmaf(f, lds[OFF_C1 + tid], i * g);
            lds[OFF_C1 + tid] = c;
            lds[OFF_H1 + tid] = o * tanh_fast(c);
        }
        __syncthreads();

        // layer 2: z[j] = b2[j] + Wih2[j,:] . h1 + Whh2[j,:] . h2
        {
            float acc = lds[OFF_B2 + tid];
            acc += dot_row<16>(&lds[OFF_WIH2 + tid * 68], &lds[OFF_H1]);
            acc += dot_row<16>(&lds[OFF_WHH2 + tid * 68], &lds[OFF_H2]);
            lds[OFF_Z + tid] = acc;
        }
        __syncthreads();
        if (tid < HID) {
            const float i = sigm(lds[OFF_Z + tid]);
            const float f = sigm(lds[OFF_Z + HID + tid]);
            const float g = tanh_fast(lds[OFF_Z + 2 * HID + tid]);
            const float o = sigm(lds[OFF_Z + 3 * HID + tid]);
            const float c = fmaf(f, lds[OFF_C2 + tid], i * g);
            lds[OFF_C2 + tid] = c;
            lds[OFF_H2 + tid] = o * tanh_fast(c);
        }
        __syncthreads();

        // layer 3: z[j] = b3[j] + Wih3[j,:] . h2 + Whh3[j,:] . h3   (152 rows)
        if (tid < G3) {
            float acc = lds[OFF_B3 + tid];
            acc += dot_row<16>(&lds[OFF_WIH3 + tid * 68], &lds[OFF_H2]);
            acc += dot_row<10>(&lds[OFF_WHH3 + tid * 40], &lds[OFF_H3]);
            lds[OFF_Z + tid] = acc;
        }
        __syncthreads();
        if (tid < FEATS) {
            const float i = sigm(lds[OFF_Z + tid]);
            const float f = sigm(lds[OFF_Z + FEATS + tid]);
            const float g = tanh_fast(lds[OFF_Z + 2 * FEATS + tid]);
            const float o = sigm(lds[OFF_Z + 3 * FEATS + tid]);
            const float c = fmaf(f, lds[OFF_C3 + tid], i * g);
            const float h = o * tanh_fast(c);
            lds[OFF_C3 + tid] = c;
            lds[OFF_H3 + tid] = h;
            out[(size_t)t * FEATS + tid] = 2.0f * h;
        }
        __syncthreads();
    }
}

// ================================ launcher ==================================
extern "C" void kernel_launch(void* const* d_in, const int* in_sizes, int n_in,
                              void* d_out, int out_size, void* d_ws, size_t ws_size,
                              hipStream_t stream) {
    const float* x    = (const float*)d_in[0];
    const float* h1   = (const float*)d_in[1];
    const float* c1   = (const float*)d_in[2];
    const float* h2   = (const float*)d_in[3];
    const float* c2   = (const float*)d_in[4];
    const float* h3   = (const float*)d_in[5];
    const float* c3   = (const float*)d_in[6];
    const float* Wih1 = (const float*)d_in[7];
    const float* Whh1 = (const float*)d_in[8];
    const float* bih1 = (const float*)d_in[9];
    const float* bhh1 = (const float*)d_in[10];
    const float* Wih2 = (const float*)d_in[11];
    const float* Whh2 = (const float*)d_in[12];
    const float* bih2 = (const float*)d_in[13];
    const float* bhh2 = (const float*)d_in[14];
    const float* Wih3 = (const float*)d_in[15];
    const float* Whh3 = (const float*)d_in[16];
    const float* bih3 = (const float*)d_in[17];
    const float* bhh3 = (const float*)d_in[18];
    float* out = (float*)d_out;

    const int T = in_sizes[0] / FEATS;
    float* pre1 = (float*)d_ws;   // T * 256 floats of scratch

    // Opt in to >64KB dynamic LDS (idempotent, not a stream op).
    (void)hipFuncSetAttribute(reinterpret_cast<const void*>(lstm_seq),
                              hipFuncAttributeMaxDynamicSharedMemorySize,
                              LDS_FLOATS * (int)sizeof(float));

    const int nblk = (T + 15) / 16;
    pre1_gemm_wmma<<<nblk, 256, 0, stream>>>(x, Wih1, bih1, bhh1, pre1, T);
    lstm_seq<<<1, 256, LDS_FLOATS * sizeof(float), stream>>>(
        pre1, h1, c1, h2, c2, h3, c3,
        Whh1, Wih2, Whh2, bih2, bhh2,
        Wih3, Whh3, bih3, bhh3, out, T);
}
